// MoELayer_7507602834060
// MI455X (gfx1250) — matrix operand order repaired and checked by reading, not compiled, not run on hardware
//
#include <hip/hip_runtime.h>
#include <hip/hip_bf16.h>
#include <cstdint>

#define B_N 8192
#define D_N 1024
#define H_N 2048
#define E_N 8
#define K_N 2
#define HC  512   // H chunk held in LDS per iteration

typedef __attribute__((ext_vector_type(16))) __bf16 v16bf;
typedef __attribute__((ext_vector_type(8)))  float  v8f;

union FragAB { v16bf v; uint32_t u[8]; };

__device__ __forceinline__ v8f v8f_zero() {
  v8f z;
  #pragma unroll
  for (int i = 0; i < 8; ++i) z[i] = 0.f;
  return z;
}

__device__ __forceinline__ v8f wmma_bf16(v16bf a, v16bf b, v8f c) {
  // D = A(16x32 bf16) x B(32x16 bf16) + C(16x16 f32)
  return __builtin_amdgcn_wmma_f32_16x16x32_bf16(false, a, false, b, (short)0, c,
                                                 false, false);
}

// A fragment, row-major MxK source; this lane's row pointer passed in.
// ISA layout: lane M = lane&15; dword i holds K = (i<4 ? 2i : 2i+8) + 8*(lane>>4).
__device__ __forceinline__ v16bf load_frag_a(const __bf16* rowp, int k0, int hi) {
  FragAB f;
  const __bf16* p = rowp + k0 + 8 * hi;
  #pragma unroll
  for (int i = 0; i < 8; ++i) {
    int k = (i < 4) ? 2 * i : 2 * i + 8;
    f.u[i] = *(const uint32_t*)(p + k);
  }
  return f.v;
}

// B fragment from N-major storage BT[n][k] (stride ldk).
// ISA layout: lane N = lane&15; dword i holds K = (i<4 ? 2i : 2i+8) + 8*(lane>>4).
__device__ __forceinline__ v16bf load_frag_b(const __bf16* bt, int n, int hi,
                                             int k0, int ldk) {
  FragAB f;
  const __bf16* p = bt + (size_t)n * ldk + k0 + 8 * hi;
  #pragma unroll
  for (int i = 0; i < 8; ++i) f.u[i] = *(const uint32_t*)(p + ((i < 4) ? 2 * i : 2 * i + 8));
  return f.v;
}

// ---------------------------------------------------------------- init
__global__ void init_kernel(int* counts) {
  if (threadIdx.x < E_N) counts[threadIdx.x] = 0;
}

// ---------------------------------------------------------------- gating
// One wave32 per token: logits -> softmax -> top2 -> renorm.
// Also converts x row to bf16 and emits per-block prob partials.
__global__ __launch_bounds__(256) void gate_kernel(
    const float* __restrict__ x, const float* __restrict__ gw,
    const float* __restrict__ gb, __bf16* __restrict__ xb,
    int* __restrict__ topk_idx, float* __restrict__ topk_w,
    int* __restrict__ counts, float* __restrict__ gate_part) {
  __shared__ float sp[8][E_N];
  const int wid = threadIdx.x >> 5, lane = threadIdx.x & 31;
  const int t = blockIdx.x * 8 + wid;

  float acc[E_N];
  #pragma unroll
  for (int e = 0; e < E_N; ++e) acc[e] = 0.f;

  const float* xr = x + (size_t)t * D_N;
  for (int d = lane; d < D_N; d += 32) {
    float xv = xr[d];
    xb[(size_t)t * D_N + d] = (__bf16)xv;
    #pragma unroll
    for (int e = 0; e < E_N; ++e) acc[e] += xv * gw[d * E_N + e];
  }
  #pragma unroll
  for (int off = 16; off > 0; off >>= 1) {
    #pragma unroll
    for (int e = 0; e < E_N; ++e) acc[e] += __shfl_xor(acc[e], off, 32);
  }

  if (lane == 0) {
    float lg[E_N], mx = -1e30f;
    #pragma unroll
    for (int e = 0; e < E_N; ++e) { lg[e] = acc[e] + gb[e]; mx = fmaxf(mx, lg[e]); }
    float s = 0.f;
    #pragma unroll
    for (int e = 0; e < E_N; ++e) { lg[e] = expf(lg[e] - mx); s += lg[e]; }
    const float inv = 1.f / s;
    #pragma unroll
    for (int e = 0; e < E_N; ++e) { lg[e] *= inv; sp[wid][e] = lg[e]; }
    int i0 = 0; float p0 = lg[0];
    #pragma unroll
    for (int e = 1; e < E_N; ++e) if (lg[e] > p0) { p0 = lg[e]; i0 = e; }
    int i1 = -1; float p1 = -1.f;
    #pragma unroll
    for (int e = 0; e < E_N; ++e)
      if (e != i0 && lg[e] > p1) { p1 = lg[e]; i1 = e; }
    const float rn = 1.f / (p0 + p1);
    topk_idx[2 * t] = i0;     topk_idx[2 * t + 1] = i1;
    topk_w[2 * t] = p0 * rn;  topk_w[2 * t + 1] = p1 * rn;
    atomicAdd(&counts[i0], 1);
    atomicAdd(&counts[i1], 1);
  }
  __syncthreads();
  if (threadIdx.x < E_N) {
    float s = 0.f;
    #pragma unroll
    for (int w = 0; w < 8; ++w) s += sp[w][threadIdx.x];
    gate_part[blockIdx.x * E_N + threadIdx.x] = s;   // deterministic partials
  }
}

// ---------------------------------------------------------------- weight convert+transpose
// src [E][R][C] f32  ->  dst [E][C][R] bf16 (N-major for B-fragment loads)
__global__ __launch_bounds__(256) void convT_kernel(const float* __restrict__ src,
                                                    __bf16* __restrict__ dst,
                                                    int R, int C) {
  __shared__ float tile[32][33];
  const int e = blockIdx.z;
  const int r0 = blockIdx.y * 32, c0 = blockIdx.x * 32;
  const float* s = src + (size_t)e * R * C;
  __bf16* dd = dst + (size_t)e * R * C;
  const int tx = threadIdx.x, ty = threadIdx.y;  // 32 x 8
  #pragma unroll
  for (int i = 0; i < 4; ++i)
    tile[ty + i * 8][tx] = s[(size_t)(r0 + ty + i * 8) * C + c0 + tx];
  __syncthreads();
  #pragma unroll
  for (int i = 0; i < 4; ++i)
    dd[(size_t)(c0 + ty + i * 8) * R + r0 + tx] = (__bf16)tile[tx][ty + i * 8];
}

// ---------------------------------------------------------------- scan: lb loss + offsets
__global__ void scan_kernel(const float* __restrict__ gate_part,
                            const int* __restrict__ counts, int* __restrict__ offs,
                            int* __restrict__ cursor, float* __restrict__ lb_out,
                            int nblk) {
  __shared__ float us[E_N];
  const int tid = threadIdx.x;
  if (tid < E_N) {
    float s = 0.f;
    for (int b = 0; b < nblk; ++b) s += gate_part[b * E_N + tid];
    us[tid] = s / (float)B_N;
  }
  __syncthreads();
  if (tid == 0) {
    float lb = 0.f;
    #pragma unroll
    for (int e = 0; e < E_N; ++e) { float d = us[e] - 1.f / E_N; lb += d * d; }
    lb_out[0] = 0.01f * lb;
    int o = 0;
    for (int e = 0; e < E_N; ++e) { offs[e] = o; cursor[e] = o; o += counts[e]; }
  }
}

// ---------------------------------------------------------------- scatter tokens to slots
__global__ __launch_bounds__(256) void scatter_kernel(const int* __restrict__ topk_idx,
                                                      int* __restrict__ cursor,
                                                      int* __restrict__ row_token,
                                                      int* __restrict__ slot_of) {
  const int i = blockIdx.x * blockDim.x + threadIdx.x;  // < B*K
  const int e = topk_idx[i];
  const int pos = atomicAdd(&cursor[e], 1);
  row_token[pos] = i >> 1;
  slot_of[i] = pos;
}

// ---------------------------------------------------------------- grouped expert FFN (WMMA)
// Block: 32 gathered rows x full D output, 16 waves (512 thr). H in HC chunks:
//   Phase A: hbuf[32][HC] = relu(X*W1_chunk + b1)   (bf16 in LDS, 32 KB)
//            wave owns 32 h-cols/chunk -> 4 tiles
//   Phase B: Y(8 tiles/wave = 64 VGPRs, f32) += hbuf * W2_chunk
//            wave owns 64 out-cols -> 8 tiles x 16 k-steps
__global__ __launch_bounds__(512) void moe_gemm_kernel(
    const __bf16* __restrict__ xb,
    const __bf16* __restrict__ w1b,   // [E][H][D] bf16 (N-major)
    const __bf16* __restrict__ w2b,   // [E][D][H] bf16 (N-major)
    const float* __restrict__ b1, const float* __restrict__ b2,
    const int* __restrict__ row_token, const int* __restrict__ counts,
    const int* __restrict__ offs, float* __restrict__ y) {
  __shared__ __bf16 hbuf[32 * HC];

  const int e = blockIdx.y;
  const int cnt = counts[e];
  const int tile = blockIdx.x;
  if (tile * 32 >= cnt) return;                     // block-uniform
  const int s0 = offs[e] + tile * 32;
  const int smax = offs[e] + cnt - 1;
  const int valid = min(32, cnt - tile * 32);

  const int w = threadIdx.x >> 5;                   // 0..15
  const int lane = threadIdx.x & 31;
  const int lo = lane & 15;
  const int hi = lane >> 4;

  // gathered A rows for this lane (rows lo and 16+lo of the tile)
  const __bf16* arow0 = xb + (size_t)row_token[min(s0 + lo, smax)] * D_N;
  const __bf16* arow1 = xb + (size_t)row_token[min(s0 + 16 + lo, smax)] * D_N;

  const __bf16* w1e = w1b + (size_t)e * H_N * D_N;
  const __bf16* w2e = w2b + (size_t)e * D_N * H_N;
  const float* b1e = b1 + (size_t)e * H_N;
  const float* b2e = b2 + (size_t)e * D_N;

  v8f Y[8];
  #pragma unroll
  for (int i = 0; i < 8; ++i) Y[i] = v8f_zero();

  for (int c = 0; c < H_N / HC; ++c) {
    // -------- Phase A: h chunk; wave owns 32 cols (2 N tiles x 2 M tiles)
    #pragma unroll
    for (int mt = 0; mt < 2; ++mt) {
      const __bf16* arow = mt ? arow1 : arow0;
      #pragma unroll
      for (int ntl = 0; ntl < 2; ++ntl) {
        const int ncol = c * HC + w * 32 + ntl * 16;  // global h column base
        v8f acc = v8f_zero();
        for (int k0 = 0; k0 < D_N; k0 += 32) {
          v16bf a = load_frag_a(arow, k0, hi);
          v16bf b = load_frag_b(w1e, ncol + lo, hi, k0, D_N);
          acc = wmma_bf16(a, b, acc);
        }
        const float bias = b1e[ncol + lo];
        const int colL = w * 32 + ntl * 16 + lo;      // column within chunk
        #pragma unroll
        for (int r = 0; r < 8; ++r) {
          float v = acc[r] + bias;
          v = v > 0.f ? v : 0.f;                       // relu
          hbuf[(mt * 16 + r + 8 * hi) * HC + colL] = (__bf16)v;
        }
      }
    }
    __syncthreads();
    // -------- Phase B: Y += h_chunk @ W2_chunk (wave owns 64 out cols)
    #pragma unroll
    for (int mt = 0; mt < 2; ++mt) {
      #pragma unroll
      for (int nt = 0; nt < 4; ++nt) {
        v8f acc = Y[mt * 4 + nt];
        const int n0 = w * 64 + nt * 16;
        for (int k0 = 0; k0 < HC; k0 += 32) {
          v16bf a = load_frag_a(&hbuf[(mt * 16 + lo) * HC], k0, hi);
          v16bf b = load_frag_b(w2e, n0 + lo, hi, c * HC + k0, H_N);
          acc = wmma_bf16(a, b, acc);
        }
        Y[mt * 4 + nt] = acc;
      }
    }
    __syncthreads();
  }

  // -------- epilogue: + b2, scatter rows to y[slot]
  #pragma unroll
  for (int mt = 0; mt < 2; ++mt) {
    #pragma unroll
    for (int nt = 0; nt < 4; ++nt) {
      const int col = w * 64 + nt * 16 + lo;
      const float bias = b2e[col];
      #pragma unroll
      for (int r = 0; r < 8; ++r) {
        const int m = mt * 16 + r + 8 * hi;
        if (m < valid)
          y[(size_t)(s0 + m) * D_N + col] = Y[mt * 4 + nt][r] + bias;
      }
    }
  }
}

// ---------------------------------------------------------------- weighted combine
__global__ __launch_bounds__(256) void combine_kernel(const float* __restrict__ y,
                                                      const float* __restrict__ topk_w,
                                                      const int* __restrict__ slot_of,
                                                      float* __restrict__ out) {
  const int t = blockIdx.x;
  const int d = threadIdx.x * 4;
  const float w0 = topk_w[2 * t], w1 = topk_w[2 * t + 1];
  const int p0 = slot_of[2 * t], p1 = slot_of[2 * t + 1];
  const float4 a = *(const float4*)(y + (size_t)p0 * D_N + d);
  const float4 b = *(const float4*)(y + (size_t)p1 * D_N + d);
  float4 o;
  o.x = w0 * a.x + w1 * b.x;
  o.y = w0 * a.y + w1 * b.y;
  o.z = w0 * a.z + w1 * b.z;
  o.w = w0 * a.w + w1 * b.w;
  *(float4*)(out + (size_t)t * D_N + d) = o;
}

// ---------------------------------------------------------------- host
extern "C" void kernel_launch(void* const* d_in, const int* in_sizes, int n_in,
                              void* d_out, int out_size, void* d_ws, size_t ws_size,
                              hipStream_t stream) {
  const float* x  = (const float*)d_in[0];
  const float* gw = (const float*)d_in[1];
  const float* gb = (const float*)d_in[2];
  const float* w1 = (const float*)d_in[3];
  const float* b1 = (const float*)d_in[4];
  const float* w2 = (const float*)d_in[5];
  const float* b2 = (const float*)d_in[6];
  float* out = (float*)d_out;

  char* ws = (char*)d_ws;
  size_t off = 0;
  auto take = [&](size_t bytes) -> char* {
    char* p = ws + off;
    off = (off + bytes + 255) & ~(size_t)255;
    return p;
  };
  __bf16* xb        = (__bf16*)take((size_t)B_N * D_N * 2);          // 16 MB
  __bf16* w1b       = (__bf16*)take((size_t)E_N * D_N * H_N * 2);    // 32 MB
  __bf16* w2b       = (__bf16*)take((size_t)E_N * D_N * H_N * 2);    // 32 MB
  float*  ybuf      = (float*) take((size_t)B_N * K_N * D_N * 4);    // 64 MB
  int*    topk_idx  = (int*)   take((size_t)B_N * K_N * 4);
  float*  topk_w    = (float*) take((size_t)B_N * K_N * 4);
  int*    slot_of   = (int*)   take((size_t)B_N * K_N * 4);
  int*    row_token = (int*)   take((size_t)B_N * K_N * 4);
  float*  gate_part = (float*) take((size_t)(B_N / 8) * E_N * 4);
  int*    counts    = (int*)   take(64);
  int*    offsb     = (int*)   take(64);
  int*    cursor    = (int*)   take(64);
  (void)ws_size; (void)in_sizes; (void)n_in; (void)out_size;

  init_kernel<<<1, 32, 0, stream>>>(counts);
  gate_kernel<<<B_N / 8, 256, 0, stream>>>(x, gw, gb, xb, topk_idx, topk_w,
                                           counts, gate_part);
  convT_kernel<<<dim3(H_N / 32, D_N / 32, E_N), dim3(32, 8), 0, stream>>>(
      w1, w1b, D_N, H_N);  // -> [E][H][D]
  convT_kernel<<<dim3(D_N / 32, H_N / 32, E_N), dim3(32, 8), 0, stream>>>(
      w2, w2b, H_N, D_N);  // -> [E][D][H]
  scan_kernel<<<1, 32, 0, stream>>>(gate_part, counts, offsb, cursor,
                                    out + (size_t)B_N * D_N, B_N / 8);
  scatter_kernel<<<(B_N * K_N) / 256, 256, 0, stream>>>(topk_idx, cursor,
                                                        row_token, slot_of);
  moe_gemm_kernel<<<dim3(B_N / 32, E_N), 512, 0, stream>>>(
      xb, w1b, w2b, b1, b2, row_token, counts, offsb, ybuf);
  combine_kernel<<<B_N, 256, 0, stream>>>(ybuf, topk_w, slot_of, out);
}
